// NCEAverage_18279380812338
// MI455X (gfx1250) — matrix operand order repaired and checked by reading, not compiled, not run on hardware
//
#include <hip/hip_runtime.h>
#include <math.h>

#define BDIM 128
#define DDIM 128
#define CDIM 24
#define POSN 128
#define NEGN 4096
#define KTOT (POSN + NEGN)
#define BSTR (DDIM + 8)   // B-table row stride in halfs: 272B (16B aligned, bank-spread)

typedef __attribute__((ext_vector_type(16))) _Float16 v16h;
typedef __attribute__((ext_vector_type(8)))  float    v8f;

// ---------------------------------------------------------------------------
// Phase 0: per-memory-slot class bitmask (one-hot label_memory -> 24-bit mask)
// ---------------------------------------------------------------------------
__global__ void mask_kernel(const float* __restrict__ label_memory,
                            unsigned* __restrict__ mask, int N) {
    int n = blockIdx.x * blockDim.x + threadIdx.x;
    if (n >= N) return;
    const float* row = label_memory + (size_t)n * CDIM;
    unsigned m = 0u;
#pragma unroll
    for (int c = 0; c < CDIM; ++c)
        m |= (row[c] > 0.0f) ? (1u << c) : 0u;
    mask[n] = m;
}

// ---------------------------------------------------------------------------
// Phase 1: per-sample ordered stream compaction over idx_total:
// first POSN positives and first NEGN negatives (in idx_total order), with
// cyclic wrap, written to sel[b][0..KTOT).
// ---------------------------------------------------------------------------
__global__ void select_kernel(const unsigned* __restrict__ mask,
                              const int* __restrict__ idx_total,
                              const int* __restrict__ y,
                              int* __restrict__ sel, int N) {
    const int b    = blockIdx.x;
    const int tid  = threadIdx.x;      // 256 threads, 8 waves of 32
    const int lane = tid & 31;
    const int wave = tid >> 5;

    __shared__ int pf[256], nf[256];
    __shared__ int wp[8], wn[8];
    __shared__ int basePos, baseNeg;
    __shared__ unsigned mb;

    if (tid == 0) { basePos = 0; baseNeg = 0; mb = mask[y[b]]; }
    __syncthreads();

    int* selb = sel + (size_t)b * KTOT;
    const int nchunks = (N + 255) / 256;

    for (int ch = 0; ch < nchunks; ++ch) {
        int j = ch * 256 + tid;
        int p = 0, ng = 0, val = 0;
        if (j < N) {
            val = idx_total[j];
            unsigned m = mask[val];
            p  = ((m & mb) != 0u) ? 1 : 0;
            ng = 1 - p;
        }
        pf[tid] = p; nf[tid] = ng;
        __syncthreads();

        // rank within wave (stable, lane order)
        int rp = 0, rn = 0;
        const int wbase = wave * 32;
        for (int i = 0; i < lane; ++i) { rp += pf[wbase + i]; rn += nf[wbase + i]; }
        if (lane == 31) { wp[wave] = rp + p; wn[wave] = rn + ng; }
        __syncthreads();

        int op = 0, on = 0, tp = 0, tn = 0;
#pragma unroll
        for (int w = 0; w < 8; ++w) {
            int a = wp[w], c = wn[w];
            if (w < wave) { op += a; on += c; }
            tp += a; tn += c;
        }
        if (p)  { int pos = basePos + op + rp; if (pos < POSN) selb[pos] = val; }
        if (ng) { int pos = baseNeg + on + rn; if (pos < NEGN) selb[POSN + pos] = val; }
        __syncthreads();
        if (tid == 0) { basePos += tp; baseNeg += tn; }
        // visibility of basePos update covered by next iteration's first barrier
        __syncthreads();
    }

    int nPos = basePos;
    int nNeg = N - nPos;
    if (nPos == 0) {                       // argmax fallback (practically unreachable)
        if (tid == 0) selb[0] = idx_total[0];
        nPos = 1;
    }
    if (nNeg <= 0) nNeg = 1;
    __threadfence_block();
    __syncthreads();

    // cyclic wrap fill (no-op when counts exceed the limits, the common case)
    for (int t = tid; t < POSN; t += 256)
        if (t >= nPos) selb[t] = selb[t % nPos];
    for (int t = tid; t < NEGN; t += 256)
        if (t >= nNeg) selb[POSN + t] = selb[POSN + (t % nNeg)];
}

// ---------------------------------------------------------------------------
// Phase 2: WMMA batched sign(memory[sel]) . {l, ab}
// Each wave: 16 K-rows x 16 cols (col0 = l, col1 = ab, cols>=2 zero),
// K-loop of 4 over D=128 with v_wmma_f32_16x16x32_f16.
// B fragments come from a zero-padded LDS table indexed by lane -> pure
// ds_load_b128, no selects, no divergent control flow in the hot loop.
// ---------------------------------------------------------------------------
// memory entries are +-1/sqrt(D), never zero: sign -> 0x3C00 | (float sign bit)
__device__ __forceinline__ _Float16 sgn_h(float x) {
    union { unsigned short u; _Float16 h; } cv;
    cv.u = (unsigned short)(0x3C00u | ((__float_as_uint(x) >> 16) & 0x8000u));
    return cv.h;
}

__global__ void __launch_bounds__(256)
wmma_kernel(const float* __restrict__ memory,
            const int*   __restrict__ sel,
            const float* __restrict__ l,
            const float* __restrict__ ab,
            float* __restrict__ out_l,
            float* __restrict__ out_ab,
            float invT) {
    const int b  = blockIdx.y;
    const int k0 = blockIdx.x * 128;          // 8 waves * 16 rows
    const int tid = threadIdx.x;

    // B-matrix table: row c holds column c of the 16-wide B operand over all
    // of D. Row 0 = l, row 1 = ab, rows 2..15 = zero.
    __shared__ _Float16 bmat[16 * BSTR];

    for (int t = tid; t < 16 * BSTR; t += 256) bmat[t] = (_Float16)0.0f;
    __syncthreads();
    if (tid < DDIM) bmat[0 * BSTR + tid]          = (_Float16)l[b * DDIM + tid];
    else            bmat[1 * BSTR + (tid - DDIM)] = (_Float16)ab[b * DDIM + (tid - DDIM)];
    __syncthreads();

    const int lane = tid & 31;
    const int wave = tid >> 5;
    const int rowM = lane & 15;               // A-row / B-col index for this lane
    const int hi   = lane >> 4;               // K-half selector

    const int krow = k0 + wave * 16 + rowM;
    const int srow = sel[(size_t)b * KTOT + krow];
    const float* mrow = memory + (size_t)srow * DDIM;

    const _Float16* brow = bmat + rowM * BSTR;

    v8f acc = {};
#pragma unroll
    for (int kb = 0; kb < DDIM; kb += 32) {
        // A fragment: contiguous K order, matching B below.
        // lanes 0-15: K = kb+0..15; lanes 16-31: K = kb+16..31.
        v16h a;
        const float* p = mrow + kb + hi * 16;
#pragma unroll
        for (int i = 0; i < 8; ++i) a[i]     = sgn_h(p[i]);
#pragma unroll
        for (int i = 0; i < 8; ++i) a[8 + i] = sgn_h(p[8 + i]);

        // B fragment: lane holds col rowM; lanes 0-15 K = kb+0..15,
        // lanes 16-31 K = kb+16..31. Straight contiguous LDS reads.
        v16h bf;
        const int koff = kb + hi * 16;
#pragma unroll
        for (int i = 0; i < 16; ++i) bf[i] = brow[koff + i];

        acc = __builtin_amdgcn_wmma_f32_16x16x32_f16(
            /*neg_a=*/false, a, /*neg_b=*/false, bf,
            /*c_mod=*/(short)0, acc, /*reuse_a=*/false, /*reuse_b=*/false);
    }

    // C layout: VGPR r, lane L -> row = r + (L>=16 ? 8 : 0), col = L&15.
    const int col = lane & 15;
    if (col < 2) {
        float* op = ((col == 0) ? out_l : out_ab)
                  + (size_t)b * KTOT + k0 + wave * 16 + hi * 8;
#pragma unroll
        for (int r = 0; r < 8; ++r) op[r] = acc[r] * invT;
    }
}

// ---------------------------------------------------------------------------
// Phase 3: bulk copy (float4) + momentum row update
// ---------------------------------------------------------------------------
__global__ void copy4_kernel(const float4* __restrict__ src,
                             float4* __restrict__ dst, size_t n4) {
    size_t i = (size_t)blockIdx.x * blockDim.x + threadIdx.x;
    size_t stride = (size_t)gridDim.x * blockDim.x;
    for (; i < n4; i += stride) dst[i] = src[i];
}

__global__ void update_kernel(const float* __restrict__ l,
                              const float* __restrict__ ab,
                              const float* __restrict__ memory,
                              const float* __restrict__ labels,
                              const int* __restrict__ y,
                              float* __restrict__ new_memory,
                              float* __restrict__ new_label_memory) {
    const int b = blockIdx.x;
    const int d = threadIdx.x;                // 128 threads, 4 waves
    const int lane = d & 31, wave = d >> 5;
    __shared__ float ws1[4], ws2[4];

    const int yb = y[b];
    float v = 0.5f * (l[b * DDIM + d] + ab[b * DDIM + d]);

    float s = v * v;
#pragma unroll
    for (int o = 16; o > 0; o >>= 1) s += __shfl_xor(s, o, 32);
    if (lane == 0) ws1[wave] = s;
    __syncthreads();
    float n1 = sqrtf(ws1[0] + ws1[1] + ws1[2] + ws1[3]);
    float vn = v / n1;

    float lp = memory[(size_t)yb * DDIM + d] * 0.5f + vn * 0.5f;
    float s2 = lp * lp;
#pragma unroll
    for (int o = 16; o > 0; o >>= 1) s2 += __shfl_xor(s2, o, 32);
    if (lane == 0) ws2[wave] = s2;
    __syncthreads();
    float n2 = sqrtf(ws2[0] + ws2[1] + ws2[2] + ws2[3]);

    new_memory[(size_t)yb * DDIM + d] = lp / n2;
    if (d < CDIM)
        new_label_memory[(size_t)yb * CDIM + d] = labels[b * CDIM + d];
}

// ---------------------------------------------------------------------------
extern "C" void kernel_launch(void* const* d_in, const int* in_sizes, int n_in,
                              void* d_out, int out_size, void* d_ws, size_t ws_size,
                              hipStream_t stream) {
    (void)n_in; (void)out_size; (void)ws_size;
    const float* labels       = (const float*)d_in[0];
    const float* l            = (const float*)d_in[1];
    const float* ab           = (const float*)d_in[2];
    const int*   y            = (const int*)d_in[3];
    const int*   idx_total    = (const int*)d_in[4];
    const float* memory       = (const float*)d_in[5];
    const float* label_memory = (const float*)d_in[6];
    const int N = in_sizes[4];

    float* out              = (float*)d_out;
    float* out_l            = out;                              // [B, KTOT, 1]
    float* out_ab           = out_l + (size_t)BDIM * KTOT;      // [B, KTOT, 1]
    float* new_memory       = out_ab + (size_t)BDIM * KTOT;     // [N, D]
    float* new_label_memory = new_memory + (size_t)N * DDIM;    // [N, C]

    // workspace: masks then sel, 256B aligned
    unsigned* mask = (unsigned*)d_ws;
    size_t mask_bytes = (((size_t)N * sizeof(unsigned)) + 255) & ~(size_t)255;
    int* sel = (int*)((char*)d_ws + mask_bytes);

    const float invT = 1.0f / (0.07f * sqrtf((float)DDIM));

    mask_kernel<<<(N + 255) / 256, 256, 0, stream>>>(label_memory, mask, N);
    select_kernel<<<BDIM, 256, 0, stream>>>(mask, idx_total, y, sel, N);

    dim3 grid(KTOT / 128, BDIM);
    wmma_kernel<<<grid, 256, 0, stream>>>(memory, sel, l, ab, out_l, out_ab, invT);

    size_t mem4 = (size_t)N * DDIM / 4;
    size_t lab4 = (size_t)N * CDIM / 4;
    copy4_kernel<<<2048, 256, 0, stream>>>((const float4*)memory, (float4*)new_memory, mem4);
    copy4_kernel<<<1024, 256, 0, stream>>>((const float4*)label_memory, (float4*)new_label_memory, lab4);
    update_kernel<<<BDIM, DDIM, 0, stream>>>(l, ab, memory, labels, y,
                                             new_memory, new_label_memory);
}
